// GNN_47519518162992
// MI455X (gfx1250) — matrix operand order repaired and checked by reading, not compiled, not run on hardware
//
#include <hip/hip_runtime.h>
#include <hip/hip_bf16.h>

typedef __attribute__((ext_vector_type(16))) _Float16 v16h;
typedef __attribute__((ext_vector_type(8)))  _Float16 v8h;
typedef __attribute__((ext_vector_type(4)))  _Float16 v4h;
typedef __attribute__((ext_vector_type(8)))  float    v8f;

// ---------------- problem constants (from reference) ----------------
#define BATCH   128
#define NM      100
#define NE      50
#define NS      25
#define NNODE   175          // per batch
#define HID     768
#define TDIM    256
#define DFEAT   1024
#define OUTF    768
#define NLAYER  3
#define NNTOT   (BATCH * NNODE)   // 22400
#define NEG_SLOPE 0.01f

// ---------------- tiny helper kernels ----------------
__global__ void zero_f32(float* p, long n) {
    long i = (long)blockIdx.x * blockDim.x + threadIdx.x;
    if (i < n) p[i] = 0.0f;
}

// Wt[n*K + k] = (f16) W[k*N + n]   (one-time weight transpose + convert)
__global__ void w_transpose_f16(const float* __restrict__ W, _Float16* __restrict__ Wt,
                                int K, int N) {
    long idx = (long)blockIdx.x * blockDim.x + threadIdx.x;
    if (idx < (long)K * N) {
        long k = idx / N;
        int  n = (int)(idx % N);
        Wt[(long)n * K + k] = (_Float16)W[idx];
    }
}

__global__ void deg_scatter(const int* __restrict__ idx, float* __restrict__ deg, int n) {
    int i = blockIdx.x * blockDim.x + threadIdx.x;
    if (i < n) atomicAdd(&deg[idx[i]], 1.0f);
}

__global__ void deg_to_dinv(const float* __restrict__ deg, float* __restrict__ dinv, int n) {
    int i = blockIdx.x * blockDim.x + threadIdx.x;
    if (i < n) dinv[i] = rsqrtf(fmaxf(deg[i], 1.0f));
}

// ---------------- build concatenated nodes + LayerNorm ----------------
__global__ void __launch_bounds__(256)
build_nodes_ln(const float* __restrict__ mention, const float* __restrict__ entity,
               const float* __restrict__ sent,    const float* __restrict__ type_emb,
               const float* __restrict__ ln_scale, const float* __restrict__ ln_bias,
               float* __restrict__ xout) {
    int node = blockIdx.x;                 // 0..NNTOT-1
    int b = node / NNODE;
    int i = node % NNODE;
    const float* hsrc; int t;
    if (i < NM)            { hsrc = mention + ((long)b * NM + i) * HID;        t = 0; }
    else if (i < NM + NE)  { hsrc = entity  + ((long)b * NE + (i - NM)) * HID; t = 1; }
    else                   { hsrc = sent    + ((long)b * NS + (i - NM - NE)) * HID; t = 2; }
    const float* te = type_emb + t * TDIM;

    int tid = threadIdx.x;
    float v[4];
    #pragma unroll
    for (int j = 0; j < 4; ++j) {
        int f = tid + j * 256;             // 0..1023, coalesced per j
        v[j] = (f < HID) ? hsrc[f] : te[f - HID];
    }
    __shared__ float s1[256];
    __shared__ float s2[256];
    float sum = v[0] + v[1] + v[2] + v[3];
    float sq  = v[0]*v[0] + v[1]*v[1] + v[2]*v[2] + v[3]*v[3];
    s1[tid] = sum; s2[tid] = sq;
    __syncthreads();
    for (int off = 128; off > 0; off >>= 1) {
        if (tid < off) { s1[tid] += s1[tid + off]; s2[tid] += s2[tid + off]; }
        __syncthreads();
    }
    float mu   = s1[0] * (1.0f / DFEAT);
    float var  = s2[0] * (1.0f / DFEAT) - mu * mu;
    float rstd = rsqrtf(var + 1e-5f);
    float* xo = xout + (long)node * DFEAT;
    #pragma unroll
    for (int j = 0; j < 4; ++j) {
        int f = tid + j * 256;
        xo[f] = (v[j] - mu) * rstd * ln_scale[f] + ln_bias[f];
    }
}

// ---------------- edge scatter: agg[dst] += x[src] * dinv_out[src] ----------------
__global__ void __launch_bounds__(256)
edge_scatter(const float* __restrict__ x, const float* __restrict__ dinv_out,
             const int* __restrict__ src, const int* __restrict__ dst,
             float* __restrict__ agg, int n_edges) {
    long t = (long)blockIdx.x * blockDim.x + threadIdx.x;
    long total = (long)n_edges * 256;
    if (t >= total) return;
    int e = (int)(t >> 8);
    int c = (int)(t & 255);
    int s = src[e], d = dst[e];
    float sc = dinv_out[s];
    float4 v = ((const float4*)x)[(long)s * 256 + c];
    float* o = agg + ((long)d * DFEAT + c * 4);
    atomicAdd(o + 0, v.x * sc);
    atomicAdd(o + 1, v.y * sc);
    atomicAdd(o + 2, v.z * sc);
    atomicAdd(o + 3, v.w * sc);
}

// ---------------- WMMA GEMM: C[M,N] = act( (A .* rowScale) @ Bt^T + bias ) ----------------
// A: f32 [M,K]; rowScale: per-row f32 (dinv_in) if USE_SCALE; Bt: f16 [N,K] (pre-transposed)
// block: 256 thr = 8 waves; tile 128(M) x 64(N), K-step 32; wave -> 16-row strip x 4 N-tiles.
// Software pipeline: global loads for tile i+1 issue BEFORE compute of tile i; the
// f32->f16 convert + LDS commit happens AFTER the WMMAs so s_wait_loadcnt hides
// behind matrix work. Double-buffered LDS, one barrier per K-step.
#define BM 128
#define BN 64
#define KT 32
#define LDT 40   // padded LDS row stride (halves); 80B keeps 16B alignment

template<bool USE_SCALE, bool LEAKY>
__global__ void __launch_bounds__(256)
wmma_gemm(const float* __restrict__ A, const float* __restrict__ rowScale,
          const _Float16* __restrict__ Bt, const float* __restrict__ bias,
          float* __restrict__ C, int M, int N, int K) {
    __shared__ __align__(16) _Float16 la[2][BM * LDT];
    __shared__ __align__(16) _Float16 lb[2][BN * LDT];

    int m0 = blockIdx.y * BM;
    int n0 = blockIdx.x * BN;
    int tid  = threadIdx.x;
    int wave = tid >> 5;
    int lane = tid & 31;
    int lrow = lane & 15;

    // --- hoisted staging addresses ---
    // A: 128x32 f32 = 1024 float4; thread does 4; 8 float4 per row.
    const float* aptr[4];
    float ascale[4];
    int lofsA[4];
    #pragma unroll
    for (int i = 0; i < 4; ++i) {
        int e  = i * 256 + tid;
        int r  = e >> 3;          // row in tile
        int c4 = e & 7;           // float4 index within 32-col row
        aptr[i]  = A + (long)(m0 + r) * K + c4 * 4;
        lofsA[i] = r * LDT + c4 * 4;
        if (USE_SCALE) ascale[i] = rowScale[m0 + r];
    }
    // B: 64 rows x 32 halves = 256 chunks of 8 halves; thread does 1.
    int br = tid >> 2, bc = tid & 3;
    const _Float16* bptr = Bt + (long)(n0 + br) * K + bc * 8;
    int lofsB = br * LDT + bc * 8;

    // in-flight staging registers (tile k0+KT)
    float4 ra[4];
    v8h    rb;

    auto load_regs = [&](int k0) {            // issue global loads, no waits forced
        #pragma unroll
        for (int i = 0; i < 4; ++i) ra[i] = *(const float4*)(aptr[i] + k0);
        rb = *(const v8h*)(bptr + k0);
    };
    auto commit = [&](int sbuf) {             // convert + LDS store (consumes ra/rb)
        #pragma unroll
        for (int i = 0; i < 4; ++i) {
            float4 v = ra[i];
            if (USE_SCALE) { v.x *= ascale[i]; v.y *= ascale[i]; v.z *= ascale[i]; v.w *= ascale[i]; }
            v4h h = { (_Float16)v.x, (_Float16)v.y, (_Float16)v.z, (_Float16)v.w };
            *(v4h*)&la[sbuf][lofsA[i]] = h;
        }
        *(v8h*)&lb[sbuf][lofsB] = rb;
    };

    // --- fragment read addresses (ISA 16-bit 16x32 A layout / B transposed in LDS) ---
    int arow = wave * 16 + lrow;
    int kb   = (lane < 16) ? 0 : 8;    // A: halves {kb..kb+7} and {kb+16..kb+23}
    int kb2  = (lane < 16) ? 0 : 8;    // B: halves {kb2..kb2+7} and {kb2+16..kb2+23}

    v8f acc[4] = {};

    load_regs(0);
    commit(0);
    __syncthreads();

    for (int k0 = 0; k0 < K; k0 += KT) {
        int  buf      = (k0 >> 5) & 1;
        bool has_next = (k0 + KT) < K;

        if (has_next) load_regs(k0 + KT);     // global loads in flight during WMMAs

        const _Float16* pa = &la[buf][arow * LDT + kb];
        v8h alo = *(const v8h*)(pa);
        v8h ahi = *(const v8h*)(pa + 16);
        v16h afrag = __builtin_shufflevector(alo, ahi,
                         0,1,2,3,4,5,6,7,8,9,10,11,12,13,14,15);

        #pragma unroll
        for (int t = 0; t < 4; ++t) {
            const _Float16* pb = &lb[buf][(t * 16 + lrow) * LDT + kb2];
            v8h blo = *(const v8h*)(pb);
            v8h bhi = *(const v8h*)(pb + 16);
            v16h bfrag = __builtin_shufflevector(blo, bhi,
                             0,1,2,3,4,5,6,7,8,9,10,11,12,13,14,15);
            acc[t] = __builtin_amdgcn_wmma_f32_16x16x32_f16(
                        /*neg_a=*/false, afrag, /*neg_b=*/false, bfrag,
                        /*c_mod=*/(short)0, acc[t],
                        /*reuse_a=*/false, /*reuse_b=*/false);
        }

        if (has_next) commit(buf ^ 1);        // loadcnt wait lands here, after WMMAs
        __syncthreads();
    }

    // epilogue: C/D layout — lane<16: N=lane, M=r; lane>=16: N=lane-16, M=8+r
    #pragma unroll
    for (int t = 0; t < 4; ++t) {
        int n  = n0 + t * 16 + lrow;
        float bz = bias[n];
        #pragma unroll
        for (int r = 0; r < 8; ++r) {
            int m = m0 + wave * 16 + ((lane < 16) ? r : 8 + r);
            float v = acc[t][r] + bz;
            if (LEAKY) v = (v > 0.0f) ? v : NEG_SLOPE * v;
            C[(long)m * N + n] = v;
        }
    }
}

// ---------------- entity slice: out[b][e][o] = fcout[(b*175+100+e)][o] ----------------
__global__ void slice_entities(const float* __restrict__ fcout, float* __restrict__ out) {
    long idx = (long)blockIdx.x * blockDim.x + threadIdx.x;
    long total = (long)BATCH * NE * OUTF;
    if (idx >= total) return;
    int  o  = (int)(idx % OUTF);
    long be = idx / OUTF;
    int  e  = (int)(be % NE);
    int  b  = (int)(be / NE);
    out[idx] = fcout[((long)(b * NNODE + NM + e)) * OUTF + o];
}

// ---------------- host-side orchestration ----------------
extern "C" void kernel_launch(void* const* d_in, const int* in_sizes, int n_in,
                              void* d_out, int out_size, void* d_ws, size_t ws_size,
                              hipStream_t stream) {
    const float* mention  = (const float*)d_in[0];
    const float* entity   = (const float*)d_in[1];
    const float* sent     = (const float*)d_in[2];
    const int*   edge_src = (const int*)  d_in[3];
    const int*   edge_dst = (const int*)  d_in[4];
    const float* type_emb = (const float*)d_in[5];
    const float* ln_scale = (const float*)d_in[6];
    const float* ln_bias  = (const float*)d_in[7];
    const float* gcn_w    = (const float*)d_in[8];   // [L, D, D]
    const float* gcn_b    = (const float*)d_in[9];   // [L, D]
    const float* fc_w     = (const float*)d_in[10];  // [D, OUT]
    const float* fc_b     = (const float*)d_in[11];  // [OUT]
    float* out = (float*)d_out;

    const int n_edges = in_sizes[3];

    // workspace layout (256B aligned)
    char* ws = (char*)d_ws;
    size_t off = 0;
    auto alloc = [&](size_t bytes) { char* p = ws + off; off = (off + bytes + 255) & ~(size_t)255; return p; };
    float*    X    = (float*)   alloc((size_t)NNTOT * DFEAT * 4);   // node features (ping)
    float*    AGG  = (float*)   alloc((size_t)NNTOT * DFEAT * 4);   // aggregation / fc-out (pong)
    _Float16* W16  = (_Float16*)alloc((size_t)NLAYER * DFEAT * DFEAT * 2); // gcn weights, [N][K] f16
    _Float16* FC16 = (_Float16*)alloc((size_t)OUTF * DFEAT * 2);           // fc weight,  [OUT][D] f16
    float*    degO = (float*)   alloc((size_t)NNTOT * 4);
    float*    degI = (float*)   alloc((size_t)NNTOT * 4);
    float*    dinvO= (float*)   alloc((size_t)NNTOT * 4);
    float*    dinvI= (float*)   alloc((size_t)NNTOT * 4);
    (void)ws_size;

    // 1) transpose+convert weights to f16 [N][K]
    {
        long n = (long)DFEAT * DFEAT;
        int blocks = (int)((n + 255) / 256);
        for (int l = 0; l < NLAYER; ++l)
            w_transpose_f16<<<blocks, 256, 0, stream>>>(gcn_w + (long)l * n, W16 + (long)l * n, DFEAT, DFEAT);
        long nf = (long)DFEAT * OUTF;
        w_transpose_f16<<<(int)((nf + 255) / 256), 256, 0, stream>>>(fc_w, FC16, DFEAT, OUTF);
    }

    // 2) degrees -> dinv
    zero_f32<<<(NNTOT * 2 + 255) / 256, 256, 0, stream>>>(degO, (long)NNTOT * 2); // degO,degI contiguous
    deg_scatter<<<(n_edges + 255) / 256, 256, 0, stream>>>(edge_src, degO, n_edges);
    deg_scatter<<<(n_edges + 255) / 256, 256, 0, stream>>>(edge_dst, degI, n_edges);
    deg_to_dinv<<<(NNTOT + 255) / 256, 256, 0, stream>>>(degO, dinvO, NNTOT);
    deg_to_dinv<<<(NNTOT + 255) / 256, 256, 0, stream>>>(degI, dinvI, NNTOT);

    // 3) build nodes + LayerNorm -> X
    build_nodes_ln<<<NNTOT, 256, 0, stream>>>(mention, entity, sent, type_emb, ln_scale, ln_bias, X);

    // 4) 3 GCN layers: zero AGG, scatter, WMMA GEMM (+bias, LeakyReLU) back into X
    long featN = (long)NNTOT * DFEAT;
    dim3 ggrid(DFEAT / BN, NNTOT / BM);   // 16 x 175
    for (int l = 0; l < NLAYER; ++l) {
        zero_f32<<<(int)((featN + 255) / 256), 256, 0, stream>>>(AGG, featN);
        edge_scatter<<<n_edges, 256, 0, stream>>>(X, dinvO, edge_src, edge_dst, AGG, n_edges);
        wmma_gemm<true, true><<<ggrid, 256, 0, stream>>>(
            AGG, dinvI, W16 + (long)l * DFEAT * DFEAT,
            gcn_b + (long)l * DFEAT, X, NNTOT, DFEAT, DFEAT);
    }

    // 5) FC head: AGG <- X @ fc_w + fc_b  (no activation, no row scaling)
    dim3 fgrid(OUTF / BN, NNTOT / BM);    // 12 x 175
    wmma_gemm<false, false><<<fgrid, 256, 0, stream>>>(
        X, nullptr, FC16, fc_b, AGG, NNTOT, OUTF, DFEAT);

    // 6) entity slice -> d_out
    long total = (long)BATCH * NE * OUTF;
    slice_entities<<<(int)((total + 255) / 256), 256, 0, stream>>>(AGG, out);
}
